// FirstEncoderWithAttentionAsParameters_87754771792387
// MI455X (gfx1250) — matrix operand order repaired and checked by reading, not compiled, not run on hardware
//
#include <hip/hip_runtime.h>

#define NUM_SQ 64
#define SQPT   832
#define DEPTH  256
#define DFF    1024
#define NH     4
#define DH     64
#define LN_EPS 1e-5f

typedef __attribute__((ext_vector_type(16))) _Float16 v16h;
typedef __attribute__((ext_vector_type(8)))  _Float16 h8;
typedef __attribute__((ext_vector_type(8)))  float    v8f;

#define WMMA_F16(a, b, c) \
  __builtin_amdgcn_wmma_f32_16x16x32_f16(false, (a), false, (b), (short)0, (c), false, false)

// Build a v16h A/B fragment from two contiguous 8-half (16B) chunks.
static __device__ inline v16h frag_from_halves(const _Float16* p0, const _Float16* p1) {
  union { v16h v; h8 p[2]; } u;
  u.p[0] = *(const h8*)p0;
  u.p[1] = *(const h8*)p1;
  return u.v;
}

// ---------------------------------------------------------------------------
// Kernel 0: stage W1 / W2 as f16, column-major (Wcm[j][k], k contiguous)
// so WMMA B-fragments are contiguous 32B lane reads. ~1MB total, L2-resident.
// ---------------------------------------------------------------------------
__global__ __launch_bounds__(256) void prep_weights(const float* __restrict__ W1,
                                                    const float* __restrict__ W2,
                                                    _Float16* __restrict__ W1cm,
                                                    _Float16* __restrict__ W2cm) {
  int idx = blockIdx.x * 256 + threadIdx.x;
  if (idx < DFF * DEPTH) {             // W1cm[j][k] = W1[k][j],  j<1024, k<256
    int j = idx / DEPTH, k = idx % DEPTH;
    W1cm[idx] = (_Float16)W1[(size_t)k * DFF + j];
  } else {                             // W2cm[j][k] = W2[k][j],  j<256,  k<1024
    int e = idx - DFF * DEPTH;
    int j = e / DFF, k = e % DFF;
    W2cm[e] = (_Float16)W2[(size_t)k * DEPTH + j];
  }
}

// ---------------------------------------------------------------------------
// Kernel 1: gathers + softmax + ctx = attn @ V + V   (one block per position)
// ---------------------------------------------------------------------------
__global__ __launch_bounds__(256) void attn_kernel(
    const int*   __restrict__ piece_indices,   // [N][64]
    const int*   __restrict__ abi,             // [N][4096]
    const int*   __restrict__ ksq,             // [N]
    const float* __restrict__ value_table,     // [F][256]
    const float* __restrict__ awt,             // [A][4]
    const float* __restrict__ abt,             // [169][4]
    const float* __restrict__ scale_p,         // scalar
    float*       __restrict__ ctx_ws)          // [N][64][256]
{
  __shared__ _Float16 attn[NH][64][72];   // softmaxed attention, f16, padded
  __shared__ _Float16 vT[64][72];         // vT[d][k] = V[k][h*64+d], per head
  __shared__ int s_pno[64];
  __shared__ int s_pi[64];

  const int n    = blockIdx.x;
  const int tid  = threadIdx.x;
  const int lane = tid & 31;
  const int wave = tid >> 5;
  const int lrow = lane & 15;
  const int lkhi = lane >> 4;
  const float scale = scale_p[0];
  const int myksq = ksq[n];

  if (tid < 64) {
    int pi = piece_indices[(size_t)n * 64 + tid];
    s_pi[tid]  = pi;
    s_pno[tid] = pi - myksq * SQPT;
  }
  __syncthreads();

  // ---- logits + softmax: thread = (h, q), 64-long row in registers ----
  {
    const int h = tid >> 6;
    const int q = tid & 63;
    const long base = (long)myksq * (SQPT * SQPT) + (long)s_pno[q] * SQPT;
    float r[64];
    float m = -1e30f;
#pragma unroll
    for (int k = 0; k < 64; ++k) {
      size_t idx = (size_t)(base + s_pno[k]);
      float aw = awt[idx * 4 + h];
      int   bi = abi[(size_t)n * 4096 + k * 64 + q];   // transposed bias index
      float ab = abt[(size_t)bi * 4 + h];
      float lg = (aw + ab) * scale;
      r[k] = lg;
      m = fmaxf(m, lg);
    }
    float s = 0.f;
#pragma unroll
    for (int k = 0; k < 64; ++k) { r[k] = __expf(r[k] - m); s += r[k]; }
    float inv = 1.0f / s;
#pragma unroll
    for (int k = 0; k < 64; ++k) attn[h][q][k] = (_Float16)(r[k] * inv);
  }
  __syncthreads();

  // ---- per head: gather vT, then ctx tiles via WMMA (C preloaded with V) ----
  const int qt   = wave & 3;          // q row-tile
  const int dtb  = (wave >> 2) << 1;  // two d col-tiles per wave
  const int arow = qt * 16 + lrow;

  for (int h = 0; h < NH; ++h) {
    {
      int k  = tid >> 2;
      int d0 = (tid & 3) * 16;
      const float* src = value_table + (size_t)s_pi[k] * DEPTH + h * DH + d0;
#pragma unroll
      for (int i = 0; i < 16; ++i) vT[d0 + i][k] = (_Float16)src[i];
    }
    __syncthreads();

    // A fragments (shared across both d-tiles): ks = 0,1
    v16h a0 = frag_from_halves(&attn[h][arow][lkhi * 16],      &attn[h][arow][lkhi * 16 + 8]);
    v16h a1 = frag_from_halves(&attn[h][arow][32 + lkhi * 16], &attn[h][arow][40 + lkhi * 16]);

    float* dst = ctx_ws + (size_t)n * (64 * DEPTH);
#pragma unroll
    for (int dd = 0; dd < 2; ++dd) {
      int dt   = dtb + dd;
      int dcol = dt * 16 + lrow;
      v8f c;
#pragma unroll
      for (int j = 0; j < 8; ++j)          // residual: C = V[q][d]
        c[j] = (float)vT[dcol][qt * 16 + j + lkhi * 8];
      v16h b0 = frag_from_halves(&vT[dcol][lkhi * 16],      &vT[dcol][lkhi * 16 + 8]);
      v16h b1 = frag_from_halves(&vT[dcol][32 + lkhi * 16], &vT[dcol][40 + lkhi * 16]);
      c = WMMA_F16(a0, b0, c);
      c = WMMA_F16(a1, b1, c);
#pragma unroll
      for (int j = 0; j < 8; ++j) {
        int qrow = qt * 16 + j + lkhi * 8;
        dst[(size_t)qrow * DEPTH + h * DH + dcol] = c[j];
      }
    }
    __syncthreads();
  }
}

// ---------------------------------------------------------------------------
// Kernel 2: LN1 -> FFN (tiled over DFF) -> +x -> LN2  (one block per position)
// ---------------------------------------------------------------------------
__global__ __launch_bounds__(256) void ffn_kernel(
    const float* __restrict__ ctx,
    const float* __restrict__ b1,
    const float* __restrict__ b2,
    const float* __restrict__ ln1w, const float* __restrict__ ln1b,
    const float* __restrict__ ln2w, const float* __restrict__ ln2b,
    const _Float16* __restrict__ W1cm,   // [1024][256] f16
    const _Float16* __restrict__ W2cm,   // [256][1024] f16
    float* __restrict__ out)
{
  __shared__ float    xs[64][264];     // x (post-LN1) f32, later reused for y
  __shared__ _Float16 hs[64][136];     // relu(x@W1) tile, f16
  __shared__ float redS[64][4], redQ[64][4];
  __shared__ float s_mean[64], s_rstd[64];

  const int n    = blockIdx.x;
  const int tid  = threadIdx.x;
  const int lane = tid & 31;
  const int wave = tid >> 5;
  const int lrow = lane & 15;
  const int lkhi = lane >> 4;
  const int q    = tid >> 2;
  const int part = tid & 3;

  // ---- LN1(ctx) -> xs ----
  {
    const float* src = ctx + ((size_t)n * 64 + q) * DEPTH + part * 64;
    float s = 0.f, ss = 0.f;
#pragma unroll
    for (int i = 0; i < 64; ++i) { float v = src[i]; xs[q][part * 64 + i] = v; s += v; ss += v * v; }
    redS[q][part] = s; redQ[q][part] = ss;
    __syncthreads();
    if (part == 0) {
      float S = redS[q][0] + redS[q][1] + redS[q][2] + redS[q][3];
      float Q = redQ[q][0] + redQ[q][1] + redQ[q][2] + redQ[q][3];
      float mean = S * (1.f / 256.f);
      float var  = Q * (1.f / 256.f) - mean * mean;
      s_mean[q] = mean; s_rstd[q] = rsqrtf(var + LN_EPS);
    }
    __syncthreads();
    float mean = s_mean[q], rstd = s_rstd[q];
#pragma unroll
    for (int i = 0; i < 64; ++i) {
      int c = part * 64 + i;
      xs[q][c] = (xs[q][c] - mean) * rstd * ln1w[c] + ln1b[c];
    }
  }
  __syncthreads();

  const int rt   = wave & 3;    // output row-tile
  const int half = wave >> 2;   // which half of the column tiles

  // persistent output accumulators: 8 col-tiles of 16x16, init = b2
  v8f acc[8];
#pragma unroll
  for (int ct = 0; ct < 8; ++ct) {
    float bv = b2[(half * 8 + ct) * 16 + lrow];
#pragma unroll
    for (int j = 0; j < 8; ++j) acc[ct][j] = bv;
  }

  // Hoisted x A-fragments: depend only on ks, reused by all 8 DFF tiles.
  const float* xrow = &xs[rt * 16 + lrow][0];
  v16h afrag[8];
#pragma unroll
  for (int ks = 0; ks < 8; ++ks) {
    int k0 = ks * 32 + lkhi * 16;
    union { v16h v; _Float16 hh[16]; } ua;
#pragma unroll
    for (int i = 0; i < 8; ++i) {
      ua.hh[i]     = (_Float16)xrow[k0 + i];
      ua.hh[8 + i] = (_Float16)xrow[k0 + 8 + i];
    }
    afrag[ks] = ua.v;
  }

  for (int t = 0; t < 8; ++t) {          // 8 tiles of 128 DFF columns
    // prefetch next tile's weight regions (global_prefetch_b8; no counters)
    if (t < 7) {
      __builtin_prefetch(W1cm + (size_t)(t + 1) * 128 * DEPTH + lane * 64, 0, 1);
      __builtin_prefetch(W2cm + (size_t)(half * 8) * 16 * DFF + (t + 1) * 128 + lane * 2, 0, 1);
    }
    // ---- mm1: h_tile = relu(x @ W1[:, t*128 : t*128+128] + b1) ----
    v8f hacc[4];
#pragma unroll
    for (int c = 0; c < 4; ++c) {
      float bv = b1[t * 128 + (half * 4 + c) * 16 + lrow];
#pragma unroll
      for (int j = 0; j < 8; ++j) hacc[c][j] = bv;
    }
#pragma unroll
    for (int ks = 0; ks < 8; ++ks) {     // K = 256 in steps of 32
#pragma unroll
      for (int c = 0; c < 4; ++c) {
        int jcol = t * 128 + (half * 4 + c) * 16 + lrow;
        const _Float16* wp = W1cm + (size_t)jcol * DEPTH + ks * 32 + lkhi * 16;
        v16h b = frag_from_halves(wp, wp + 8);
        hacc[c] = WMMA_F16(afrag[ks], b, hacc[c]);
      }
    }
    __syncthreads();                     // prior mm2 reads of hs complete
#pragma unroll
    for (int c = 0; c < 4; ++c) {
      int col = (half * 4 + c) * 16 + lrow;
#pragma unroll
      for (int j = 0; j < 8; ++j)
        hs[rt * 16 + j + lkhi * 8][col] = (_Float16)fmaxf(hacc[c][j], 0.f);
    }
    __syncthreads();
    // ---- mm2: out += h_tile @ W2[t*128 : t*128+128, :] ----
    const _Float16* hrow = &hs[rt * 16 + lrow][0];
#pragma unroll
    for (int ks = 0; ks < 4; ++ks) {     // K = 128 in steps of 32
      int k0 = ks * 32 + lkhi * 16;
      v16h a = frag_from_halves(hrow + k0, hrow + k0 + 8);
#pragma unroll
      for (int ct = 0; ct < 8; ++ct) {
        int jcol = (half * 8 + ct) * 16 + lrow;
        const _Float16* wp = W2cm + (size_t)jcol * DFF + t * 128 + ks * 32 + lkhi * 16;
        v16h b = frag_from_halves(wp, wp + 8);
        acc[ct] = WMMA_F16(a, b, acc[ct]);
      }
    }
  }

  // ---- residual: ffn + x ----
#pragma unroll
  for (int ct = 0; ct < 8; ++ct) {
    int col = (half * 8 + ct) * 16 + lrow;
#pragma unroll
    for (int j = 0; j < 8; ++j)
      acc[ct][j] += xs[rt * 16 + j + lkhi * 8][col];
  }
  __syncthreads();
#pragma unroll
  for (int ct = 0; ct < 8; ++ct) {       // y -> xs (reuse buffer)
    int col = (half * 8 + ct) * 16 + lrow;
#pragma unroll
    for (int j = 0; j < 8; ++j)
      xs[rt * 16 + j + lkhi * 8][col] = acc[ct][j];
  }
  __syncthreads();

  // ---- LN2 -> out ----
  {
    float s = 0.f, ss = 0.f;
#pragma unroll
    for (int i = 0; i < 64; ++i) { float v = xs[q][part * 64 + i]; s += v; ss += v * v; }
    redS[q][part] = s; redQ[q][part] = ss;
    __syncthreads();
    if (part == 0) {
      float S = redS[q][0] + redS[q][1] + redS[q][2] + redS[q][3];
      float Q = redQ[q][0] + redQ[q][1] + redQ[q][2] + redQ[q][3];
      float mean = S * (1.f / 256.f);
      float var  = Q * (1.f / 256.f) - mean * mean;
      s_mean[q] = mean; s_rstd[q] = rsqrtf(var + LN_EPS);
    }
    __syncthreads();
    float mean = s_mean[q], rstd = s_rstd[q];
    float* dst = out + ((size_t)n * 64 + q) * DEPTH;
#pragma unroll
    for (int i = 0; i < 64; ++i) {
      int c = part * 64 + i;
      dst[c] = (xs[q][c] - mean) * rstd * ln2w[c] + ln2b[c];
    }
  }
}

// ---------------------------------------------------------------------------
extern "C" void kernel_launch(void* const* d_in, const int* in_sizes, int n_in,
                              void* d_out, int out_size, void* d_ws, size_t ws_size,
                              hipStream_t stream) {
  const int*   piece_indices = (const int*)d_in[0];
  const int*   abi           = (const int*)d_in[1];
  const int*   ksq           = (const int*)d_in[2];
  const float* value_table   = (const float*)d_in[3];
  const float* awt           = (const float*)d_in[4];
  const float* abt           = (const float*)d_in[5];
  const float* attn_scale    = (const float*)d_in[6];
  const float* W1            = (const float*)d_in[7];
  const float* b1            = (const float*)d_in[8];
  const float* W2            = (const float*)d_in[9];
  const float* b2            = (const float*)d_in[10];
  const float* ln1w          = (const float*)d_in[11];
  const float* ln1b          = (const float*)d_in[12];
  const float* ln2w          = (const float*)d_in[13];
  const float* ln2b          = (const float*)d_in[14];
  float* out = (float*)d_out;

  const int N = in_sizes[2];   // ksq has one entry per position

  float*    ctx_ws = (float*)d_ws;
  _Float16* W1cm   = (_Float16*)((char*)d_ws + (size_t)N * 64 * DEPTH * sizeof(float));
  _Float16* W2cm   = W1cm + (size_t)DFF * DEPTH;

  prep_weights<<<(2 * DFF * DEPTH) / 256, 256, 0, stream>>>(W1, W2, W1cm, W2cm);
  attn_kernel<<<N, 256, 0, stream>>>(piece_indices, abi, ksq, value_table, awt, abt,
                                     attn_scale, ctx_ws);
  ffn_kernel<<<N, 256, 0, stream>>>(ctx_ws, b1, b2, ln1w, ln1b, ln2w, ln2b,
                                    W1cm, W2cm, out);
}
